// Raycaster_1142461301140
// MI455X (gfx1250) — matrix operand order repaired and checked by reading, not compiled, not run on hardware
//
#include <hip/hip_runtime.h>

typedef __attribute__((ext_vector_type(16))) _Float16 v16h;
typedef __attribute__((ext_vector_type(2)))  _Float16 h2;
typedef __attribute__((ext_vector_type(8)))  float    v8f;
typedef __attribute__((ext_vector_type(4)))  float    v4f;
typedef __attribute__((ext_vector_type(2)))  float    v2f;

#define NS    192          // samples per ray
#define HID   64           // hidden width
#define NWAVE 8            // waves (rays) per 256-thread block
#define NTILE (NS/16)      // 12 sample-tiles per ray

__device__ __forceinline__ float fast_sigmoid(float x){ return 1.0f/(1.0f+__expf(-x)); }
__device__ __forceinline__ float fast_softplus(float x){ return fmaxf(x,0.0f)+log1pf(__expf(-fabsf(x))); }

__global__ __launch_bounds__(256) void raycast_fused_kernel(
    const float* __restrict__ pos, const float* __restrict__ tv,
    const float* __restrict__ W1,  const float* __restrict__ b1,
    const float* __restrict__ W2,  const float* __restrict__ b2,
    float* __restrict__ out, int R)
{
  // Per-wave private LDS regions (all traffic is intra-wave; no barriers needed)
  __shared__ _Float16 sH[NWAVE][16*HID];   // 16-sample tile of relu(h), f16
  __shared__ v4f      sC[NWAVE][NS];       // per-sample (c0,c1,c2,alpha) packed 16B
  __shared__ float    sT[NWAVE][NS];       // t_values for this ray

  const int lane = threadIdx.x & 31;
  const int wave = threadIdx.x >> 5;
  const int ray  = blockIdx.x * NWAVE + wave;
  if (ray >= R) return;                    // uniform per wave; EXEC stays all-ones

  const int n16 = lane & 15;               // N (column) index / M (row) index
  const int hi  = lane >> 4;               // half-wave selector

  // ---------------- Preload layer-1 B operand: W1ext (4x64), b1 folded as K=3 row ----
  // B 4x16 f32 layout: lane holds column n16; VGPR0 = K(2*hi), VGPR1 = K(2*hi)+1
  v2f w1v[4];
#pragma unroll
  for (int nt = 0; nt < 4; ++nt){
    int n  = nt*16 + n16;
    int k0 = 2*hi;
    w1v[nt][0] = (k0   < 3) ? W1[k0*HID + n]     : b1[n];
    w1v[nt][1] = (k0+1 < 3) ? W1[(k0+1)*HID + n] : b1[n];
  }

  // ---------------- Preload layer-2 B operand: W2 (64x4) zero-padded to 16 cols, f16 --
  // B 32x16 f16 layout per K-chunk: lane holds column n16; K = 32*ch + 16*hi + 2*v + t
  v16h w2v[2];
#pragma unroll
  for (int ch = 0; ch < 2; ++ch){
#pragma unroll
    for (int v = 0; v < 8; ++v){
#pragma unroll
      for (int t = 0; t < 2; ++t){
        int K = 32*ch + 16*hi + 2*v + t;
        float val = (n16 < 4) ? W2[K*4 + n16] : 0.0f;
        w2v[ch][2*v+t] = (_Float16)val;
      }
    }
  }
  const float b2n = (n16 < 4) ? b2[n16] : 0.0f;

  // ---------------- t_values -> LDS (coalesced, 6 per lane) -------------------------
#pragma unroll
  for (int j = 0; j < NS/32; ++j){
    int s = lane + 32*j;
    sT[wave][s] = tv[(long)ray*NS + s];
  }

  // ---------------- MLP over 12 tiles of 16 samples ---------------------------------
  for (int tile = 0; tile < NTILE; ++tile){
    const long gs = (long)ray*NS + tile*16 + n16;   // this lane's sample (row M)

    // A 16x4 f32 layout: lanes 0-15 hold K=0,1 (x,y) ; lanes 16-31 hold K=2,3 (z,1)
    // Branchless: both loads unconditional & in-bounds, one cndmask for the 1.0.
    float la = pos[gs*3 + 2*hi];     // hi=0 -> x ; hi=1 -> z
    float lb = pos[gs*3 + 1];        // y (always valid)
    v2f a1;
    a1[0] = la;
    a1[1] = hi ? 1.0f : lb;

#pragma unroll
    for (int nt = 0; nt < 4; ++nt){
      v8f hacc = {};
      hacc = __builtin_amdgcn_wmma_f32_16x16x4_f32(
          false, a1, false, w1v[nt], (short)0, hacc, false, false);
      // D layout: lane holds column (16*nt+n16); row m = v + 8*hi.  relu -> f16 -> LDS
#pragma unroll
      for (int v = 0; v < 8; ++v){
        float hv = fmaxf(hacc[v], 0.0f);        // v_max_num_f32, no canonicalize
        sH[wave][(v + 8*hi)*HID + nt*16 + n16] = (_Float16)hv;
      }
    }

    // ---- layer 2: color_o(16x4) = relu(h)(16x64) @ W2, two K=32 chained WMMAs ----
    v8f acc = {};
#pragma unroll
    for (int ch = 0; ch < 2; ++ch){
      // A 16x32 f16 layout: lane holds row n16; VGPR v holds K pair at kb
      v16h a2;
#pragma unroll
      for (int v = 0; v < 8; ++v){
        int kb = 2*v + hi*16 + 32*ch;   // even -> dword aligned
        h2 p = *(const h2*)&sH[wave][n16*HID + kb];
        a2[2*v]   = p[0];
        a2[2*v+1] = p[1];
      }
      acc = __builtin_amdgcn_wmma_f32_16x16x32_f16(
          false, a2, false, w2v[ch], (short)0, acc, false, false);
    }
    // D layout: lane holds channel n16 (<4 valid); row m = v + 8*hi
    if (n16 < 4){
#pragma unroll
      for (int v = 0; v < 8; ++v){
        // scalar channel write into the packed v4f slot
        ((float*)&sC[wave][tile*16 + v + 8*hi])[n16] = acc[v] + b2n;
      }
    }
  }

  // ---------------- Phase A: per-sample nonlinearity + alpha (parallel, 6/lane) -----
#pragma unroll
  for (int j = 0; j < NS/32; ++j){
    int s = lane + 32*j;
    v4f o = sC[wave][s];                       // ds_load_b128
    float delta = (s < NS-1) ? (sT[wave][s+1] - sT[wave][s]) : 1e10f;
    v4f r;
    r[0] = fast_sigmoid(o[0]);
    r[1] = fast_sigmoid(o[1]);
    r[2] = fast_sigmoid(o[2]);
    r[3] = 1.0f - __expf(-fast_softplus(o[3]) * delta);   // alpha
    sC[wave][s] = r;                           // ds_store_b128
  }

  // ---------------- Phase B: sequential transmittance scan (cheap chain) ------------
  float T = 1.0f;
  float csum0 = 0.0f, csum1 = 0.0f, csum2 = 0.0f, asum = 0.0f;
  float wmax = -1.0f;
  int   cut  = 0;
  for (int s = 0; s < NS; ++s){
    v4f o = sC[wave][s];                       // ds_load_b128
    float a = o[3];
    float w = a * T;
    csum0 += w * o[0];
    csum1 += w * o[1];
    csum2 += w * o[2];
    if (s < NS-1){
      asum += w;
      if (w > wmax){ wmax = w; cut = s; }      // strict > keeps first max (argmax)
    }
    T *= fminf(1.0f, 1.0f - a + 1e-10f);
  }
  if (asum < 0.1f) cut = NS - 1;
  float depth = sT[wave][cut];

  if (lane == 0){
    out[(long)ray*3 + 0] = csum0;
    out[(long)ray*3 + 1] = csum1;
    out[(long)ray*3 + 2] = csum2;
    out[(long)3*R + ray] = asum;
    out[(long)4*R + ray] = depth;
  }
}

extern "C" void kernel_launch(void* const* d_in, const int* in_sizes, int n_in,
                              void* d_out, int out_size, void* d_ws, size_t ws_size,
                              hipStream_t stream) {
  const float* pos = (const float*)d_in[0];   // (R,S,3) f32
  const float* tv  = (const float*)d_in[1];   // (R,S)   f32
  const float* W1  = (const float*)d_in[2];   // (3,64)  f32
  const float* b1  = (const float*)d_in[3];   // (64,)   f32
  const float* W2  = (const float*)d_in[4];   // (64,4)  f32
  const float* b2  = (const float*)d_in[5];   // (4,)    f32
  // d_in[6] = include_depth (==1 in this workload; depth path compiled in)
  float* out = (float*)d_out;                 // [R*3 | R | R] concatenated

  const int R = in_sizes[1] / NS;
  const int blocks = (R + NWAVE - 1) / NWAVE;
  raycast_fused_kernel<<<blocks, 256, 0, stream>>>(pos, tv, W1, b1, W2, b2, out, R);
}
